// DifferentialAttention_51161650430155
// MI455X (gfx1250) — matrix operand order repaired and checked by reading, not compiled, not run on hardware
//
#include <hip/hip_runtime.h>
#include <hip/hip_bf16.h>

// ---------------------------------------------------------------------------
// DifferentialAttention for MI455X (gfx1250), wave32 + WMMA bf16.
//   B=8, N=2048, D=KD=64
//   out = (softmax(QK^T) masked by p > rowmean) @ V,  rowmean == 1/N exactly.
// Fused flash-style: per-row (max, sumexp) online in pass 1, masked P@V in
// pass 2. All matmuls via v_wmma_f32_16x16x32_bf16.  V chunk staging uses
// GLOBAL_LOAD_ASYNC_TO_LDS (ASYNCcnt) when the toolchain exposes it, so the
// copy overlaps the S-tile WMMAs.
// ---------------------------------------------------------------------------

typedef __bf16 bf16;
typedef __bf16 v16bf __attribute__((ext_vector_type(16)));
typedef __bf16 v8bf  __attribute__((ext_vector_type(8)));
typedef float  v8f   __attribute__((ext_vector_type(8)));
typedef int    v4i   __attribute__((ext_vector_type(4)));

#define AS_GLOBAL __attribute__((address_space(1)))
#define AS_LDS    __attribute__((address_space(3)))

#define BATCH 8
#define SEQ   2048
#define DIM   64

#if defined(__gfx1250__) && __has_builtin(__builtin_amdgcn_global_load_async_to_lds_b128)
#define USE_ASYNC_LDS 1
#else
#define USE_ASYNC_LDS 0
#endif

static __device__ __forceinline__ v8f wmma_bf16(v16bf a, v16bf b, v8f c) {
    return __builtin_amdgcn_wmma_f32_16x16x32_bf16(
        /*neg_a=*/false, a, /*neg_b=*/false, b,
        /*c_mod=*/(short)0, c, /*reuse_a=*/false, /*reuse_b=*/false);
}

static __device__ __forceinline__ v16bf cat8(v8bf lo, v8bf hi) {
    return __builtin_shufflevector(lo, hi, 0, 1, 2, 3, 4, 5, 6, 7,
                                           8, 9, 10, 11, 12, 13, 14, 15);
}

static __device__ __forceinline__ void wait_asynccnt0() {
#if __has_builtin(__builtin_amdgcn_s_wait_asynccnt)
    __builtin_amdgcn_s_wait_asynccnt(0);
#else
    asm volatile("s_wait_asynccnt 0x0" ::: "memory");
#endif
}

// ---------------------------------------------------------------------------
// Kernel 1: Q/K/V projection.  x:[B*N,64] f32  x  W:[64,64] f32 (+bias)
// -> bf16 [B*N,64] in workspace.  4 waves/block, each wave owns 16 rows.
// ---------------------------------------------------------------------------
__global__ __launch_bounds__(128) void qkv_project_kernel(
    const float* __restrict__ x,
    const float* __restrict__ Wq, const float* __restrict__ bq,
    const float* __restrict__ Wk, const float* __restrict__ bk,
    const float* __restrict__ Wv, const float* __restrict__ bv,
    bf16* __restrict__ Qb, bf16* __restrict__ Kb, bf16* __restrict__ Vb)
{
    const int tid = threadIdx.x;
    const float* Ws[3] = {Wq, Wk, Wv};
    const float* bs[3] = {bq, bk, bv};

    // W^T staged as bf16 in LDS: WT[mat][out_col][in_dim]
    __shared__ __align__(32) bf16 WT[3][64][64];
    for (int idx = tid; idx < 3 * 64 * 64; idx += 128) {
        const int mat = idx >> 12;
        const int rem = idx & 4095;
        const int i = rem >> 6;      // input dim
        const int j = rem & 63;      // output col
        WT[mat][j][i] = (bf16)Ws[mat][i * 64 + j];
    }
    __syncthreads();

    const int lane = tid & 31;
    const int wave = tid >> 5;
    const int ml = lane & 15;        // A-matrix row / D-matrix column slot
    const int h  = lane >> 4;        // lane half

    const long row0 = (long)blockIdx.x * 64 + wave * 16;
    const float* xr = x + (row0 + ml) * DIM;

    // A operands (x rows as bf16), 16x32 layout (contiguous K):
    //   elems [0..7]  = K = 32c + 16h + e
    //   elems [8..15] = K = 32c + 16h + 8 + e
    v16bf A[2];
#pragma unroll
    for (int c = 0; c < 2; ++c) {
#pragma unroll
        for (int e = 0; e < 8; ++e) {
            A[c][e]     = (bf16)xr[32 * c + 16 * h + e];
            A[c][8 + e] = (bf16)xr[32 * c + 16 * h + 8 + e];
        }
    }

    bf16* outs[3] = {Qb, Kb, Vb};
#pragma unroll
    for (int mat = 0; mat < 3; ++mat) {
        bf16* op = outs[mat];
#pragma unroll
        for (int n0t = 0; n0t < 4; ++n0t) {
            const int n = n0t * 16 + ml;           // output column
            const float bias = bs[mat][n];
            v8f acc;
#pragma unroll
            for (int r = 0; r < 8; ++r) acc[r] = bias;
#pragma unroll
            for (int c = 0; c < 2; ++c) {
                // B operand 32x16: elems e = K(16h + e) within chunk c,
                // contiguous along in_dim of WT row n -> 32B aligned load.
                const v16bf Bv = *(const v16bf*)&WT[mat][n][32 * c + 16 * h];
                acc = wmma_bf16(A[c], Bv, acc);
            }
#pragma unroll
            for (int r = 0; r < 8; ++r)
                op[(row0 + r + 8 * h) * DIM + n] = (bf16)acc[r];
        }
    }
}

// ---------------------------------------------------------------------------
// Kernel 2: fused sparse attention.  One wave handles 16 query rows.
//   Pass 1: online (rowmax, sumexp) over all 2048 keys (WMMA S-tiles).
//   Pass 2: recompute S per 32-key chunk, mask p > Z/N, P@V via WMMA.
//           V chunk staged via async global->LDS copy (overlapped).
// ---------------------------------------------------------------------------
__global__ __launch_bounds__(32) void sparse_attn_kernel(
    const bf16* __restrict__ Qb, const bf16* __restrict__ Kb,
    const bf16* __restrict__ Vb, float* __restrict__ out)
{
    const int lane = threadIdx.x & 31;
    const int ml = lane & 15;
    const int h  = lane >> 4;
    const long base = (long)blockIdx.y * SEQ;          // batch row offset
    const long qr0  = (long)blockIdx.x * 16;

#if USE_ASYNC_LDS
    __shared__ __align__(16) bf16 Vraw[32][64];        // V chunk, row-major
#else
    __shared__ __align__(32) bf16 VT[64][32];          // V chunk transposed
#endif
    __shared__ __align__(32) bf16 Plds[16][32];        // masked P tile

    // ---- Q A-operands (persistent) ----
    const bf16* qrow = Qb + (base + qr0 + ml) * DIM;
    v16bf QA[2];
#pragma unroll
    for (int c = 0; c < 2; ++c) {
        const v8bf qlo = *(const v8bf*)(qrow + 32 * c + 16 * h);
        const v8bf qhi = *(const v8bf*)(qrow + 32 * c + 16 * h + 8);
        QA[c] = cat8(qlo, qhi);
    }

    // ---- Pass 1: online rowmax / sumexp ----
    float rm[8], rz[8];
#pragma unroll
    for (int r = 0; r < 8; ++r) { rm[r] = -3.0e38f; rz[r] = 0.0f; }

    for (int kt = 0; kt < SEQ / 16; ++kt) {
        const bf16* krow = Kb + (base + kt * 16 + ml) * DIM;
        if (kt + 4 < SEQ / 16)
            __builtin_prefetch(Kb + (base + (kt + 4) * 16 + ml) * DIM, 0, 3);
        v8f s = {};
#pragma unroll
        for (int c = 0; c < 2; ++c) {
            // B = K^T tile: lane column = key j, elems = feature dims
            // 32c+16h .. +15 (contiguous, 32B aligned).
            const v16bf KB = *(const v16bf*)(krow + 32 * c + 16 * h);
            s = wmma_bf16(QA[c], KB, s);
        }
#pragma unroll
        for (int r = 0; r < 8; ++r) {
            const float sv = s[r];
            const float nm = fmaxf(rm[r], sv);
            rz[r] = rz[r] * __expf(rm[r] - nm) + __expf(sv - nm);
            rm[r] = nm;
        }
    }

    // combine (m, Z) across the 16 lanes of each half
#pragma unroll
    for (int r = 0; r < 8; ++r) {
        float mv = rm[r], zv = rz[r];
#pragma unroll
        for (int off = 1; off < 16; off <<= 1) {
            const float om = __shfl_xor(mv, off, 32);
            const float oz = __shfl_xor(zv, off, 32);
            const float nm = fmaxf(mv, om);
            zv = zv * __expf(mv - nm) + oz * __expf(om - nm);
            mv = nm;
        }
        rm[r] = mv; rz[r] = zv;
    }
    float thr[8];
#pragma unroll
    for (int r = 0; r < 8; ++r) thr[r] = rz[r] * (1.0f / (float)SEQ);

    // ---- Pass 2: masked P @ V ----
    v8f oacc[4] = {};
    for (int ch = 0; ch < SEQ / 32; ++ch) {
        const int k32 = ch * 32;
        const bf16* vrow = Vb + (base + k32 + lane) * DIM;

#if USE_ASYNC_LDS
        // Kick off async DMA of this wave's V chunk (32 rows x 128B) into
        // LDS; each lane copies its own row in 8 b128 pieces.  Tracked by
        // ASYNCcnt, overlaps with the S-tile WMMAs below.
#pragma unroll
        for (int i = 0; i < 8; ++i) {
            __builtin_amdgcn_global_load_async_to_lds_b128(
                (AS_GLOBAL v4i*)(vrow + i * 8),
                (AS_LDS v4i*)&Vraw[lane][i * 8],
                0, 0);
        }
#else
        // Synchronous fallback: stage V chunk transposed: VT[d][klocal]
#pragma unroll
        for (int q = 0; q < 4; ++q) {
            const v16bf vv = *(const v16bf*)(vrow + q * 16);
#pragma unroll
            for (int e = 0; e < 16; ++e) VT[q * 16 + e][lane] = vv[e];
        }
#endif

        // recompute S for two 16-col groups, mask, stash as bf16 P tile
#pragma unroll
        for (int g = 0; g < 2; ++g) {
            const bf16* krow = Kb + (base + k32 + g * 16 + ml) * DIM;
            v8f s = {};
#pragma unroll
            for (int c = 0; c < 2; ++c) {
                const v16bf KB = *(const v16bf*)(krow + 32 * c + 16 * h);
                s = wmma_bf16(QA[c], KB, s);
            }
#pragma unroll
            for (int r = 0; r < 8; ++r) {
                float p = __expf(s[r] - rm[r]);
                p = (p > thr[r]) ? p : 0.0f;     // mean-threshold sparsify
                Plds[r + 8 * h][g * 16 + ml] = (bf16)p;
            }
        }

#if USE_ASYNC_LDS
        wait_asynccnt0();
#endif
        __syncthreads();

        // P as A-operand (16x32)
        const v8bf plo = *(const v8bf*)&Plds[ml][16 * h];
        const v8bf phi = *(const v8bf*)&Plds[ml][16 * h + 8];
        const v16bf PA = cat8(plo, phi);

        // V as B-operand per 16-wide output tile:
        //   VB[e] = V[k32 + 16h + e][d0 + ml]
#pragma unroll
        for (int d0t = 0; d0t < 4; ++d0t) {
#if USE_ASYNC_LDS
            v16bf VB;
#pragma unroll
            for (int e = 0; e < 16; ++e)
                VB[e] = Vraw[16 * h + e][d0t * 16 + ml];
#else
            const v16bf VB = *(const v16bf*)&VT[d0t * 16 + ml][16 * h];
#endif
            oacc[d0t] = wmma_bf16(PA, VB, oacc[d0t]);
        }
        __syncthreads();
    }

    // ---- normalize by Z and store f32 ----
#pragma unroll
    for (int d0t = 0; d0t < 4; ++d0t) {
#pragma unroll
        for (int r = 0; r < 8; ++r) {
            out[(base + qr0 + r + 8 * h) * DIM + d0t * 16 + ml] =
                oacc[d0t][r] * (1.0f / rz[r]);
        }
    }
}

// ---------------------------------------------------------------------------
extern "C" void kernel_launch(void* const* d_in, const int* in_sizes, int n_in,
                              void* d_out, int out_size, void* d_ws, size_t ws_size,
                              hipStream_t stream)
{
    const float* x  = (const float*)d_in[0];
    const float* Wq = (const float*)d_in[1];
    const float* bq = (const float*)d_in[2];
    const float* Wk = (const float*)d_in[3];
    const float* bk = (const float*)d_in[4];
    const float* Wv = (const float*)d_in[5];
    const float* bv = (const float*)d_in[6];

    const size_t elems = (size_t)BATCH * SEQ * DIM;   // 1M elems
    bf16* Qb = (bf16*)d_ws;
    bf16* Kb = Qb + elems;
    bf16* Vb = Kb + elems;                            // 6 MB total bf16

    qkv_project_kernel<<<dim3(BATCH * SEQ / 64), 128, 0, stream>>>(
        x, Wq, bq, Wk, bk, Wv, bv, Qb, Kb, Vb);

    sparse_attn_kernel<<<dim3(SEQ / 16, BATCH), 32, 0, stream>>>(
        Qb, Kb, Vb, (float*)d_out);
}
